// Agent2ValueNet_21715354648582
// MI455X (gfx1250) — compile-verified, matrix operand order checked by reading, not run on hardware
//
#include <hip/hip_runtime.h>

// Fully-fused Agent2 value-net for gfx1250 (MI455X), round 2.
//
// Prep kernel: transpose+convert all conv weights once into d_ws as f16
//   Wt[n][k] (conv0 K padded 105->128). ~1.8 MB.
// Main kernel: one workgroup (256 thr = 8 wave32) processes TILE_B=16 samples
//   end-to-end. Activations live in LDS f16 (x/a/b, 3x96KB). Each wave owns a
//   16-channel N-tile; its whole per-conv B operand (12 v16h fragments) is
//   loaded once from global (pre-transposed f16) and held in VGPRs across all
//   24 M-tiles. A-fragments are implicit-im2col ds_load_b128 pairs. All convs
//   are v_wmma_f32_16x16x32_f16 with f32 accumulation.

typedef __attribute__((ext_vector_type(16))) _Float16 v16h;
typedef __attribute__((ext_vector_type(8)))  _Float16 v8h;
typedef __attribute__((ext_vector_type(8)))  float    v8f;

namespace {
constexpr int LBOARD  = 24;
constexpr int CIN     = 15;
constexpr int F       = 128;
constexpr int NB      = 9;
constexpr int TILE_B  = 16;
constexpr int MROWS   = TILE_B * LBOARD;   // 384 im2col rows per workgroup
constexpr int MT      = MROWS / 16;        // 24 M-tiles
constexpr int KW_RES  = 3 * F;             // 384 (K for res convs)
constexpr int KCH_RES = KW_RES / 32;       // 12 WMMA k-chunks
constexpr int KW_C0   = 128;               // conv0 K padded 105 -> 128
constexpr int KCH_C0  = 4;
constexpr int K0_REAL = 7 * CIN;           // 105
constexpr int THREADS = 256;
constexpr int POOLW   = 144;               // padded pooled-vector stride

// d_ws layout (f16): [conv0 128x128][18 x (128x384)]
constexpr int WS_C0    = F * KW_C0;               // 16384 halfs
constexpr int WS_RES1  = 3 * F * F;               // 49152 halfs per res conv
constexpr int WS_TOTAL = WS_C0 + 18 * WS_RES1;    // 901120 halfs (~1.8 MB)

// LDS layout (byte offsets)
constexpr int OFF_X    = 0;                       // x residual 384x128 f16
constexpr int OFF_A    = OFF_X + MROWS * F * 2;   // a scratch
constexpr int OFF_B    = OFF_A + MROWS * F * 2;   // b scratch
constexpr int OFF_LNS  = OFF_B + MROWS * F * 2;   // 128 f32
constexpr int OFF_LNB  = OFF_LNS + F * 4;         // 128 f32
constexpr int OFF_POOL = OFF_LNB + F * 4;         // 16 x 144 f32
constexpr int SMEM_BYTES = OFF_POOL + TILE_B * POOLW * 4;  // ~298 KB < 320 KB
}  // namespace

// ---------------------------------------------------------------------------
// Weight prep: f32 (K,Cin,Cout) -> f16 Wt[co][k] in d_ws.
extern "C" __global__ void __launch_bounds__(256)
prep_weights(const float* __restrict__ w0,   // (7,15,128)
             const float* __restrict__ cw1,  // (9,3,128,128)
             const float* __restrict__ cw2,  // (9,3,128,128)
             _Float16* __restrict__ ws) {
  const int idx = blockIdx.x * 256 + threadIdx.x;
  if (idx < WS_C0) {  // conv0: k = dk*15+ci -> flat k index in w0 already
    const int co = idx >> 7;
    const int k  = idx & 127;
    const float v = (k < K0_REAL) ? w0[k * F + co] : 0.0f;
    ws[co * KW_C0 + k] = (_Float16)v;
  } else {
    const int t = idx - WS_C0;
    if (t < 18 * WS_RES1) {
      const int j   = t / WS_RES1;        // conv 0..17 (blk*2 + {0,1})
      const int r   = t - j * WS_RES1;
      const int kk  = r >> 7;             // dk*128+ci
      const int co  = r & 127;
      const int blk = j >> 1;
      const float* w = (j & 1) ? cw2 : cw1;
      ws[WS_C0 + j * WS_RES1 + co * KW_RES + kk] =
          (_Float16)w[blk * WS_RES1 + kk * F + co];
    }
  }
}

// ---------------------------------------------------------------------------
// GEMM for one conv layer. 8 waves: wave w owns N-tile w (16 output channels).
// B operand loaded once from global (pre-transposed f16) into VGPRs.
// MODE 0: dst = relu(acc+bias)   (conv0)
// MODE 1: dst = acc+bias         (conv1, LN follows)
// MODE 2: dst += acc+bias        (conv2, residual add into x)
// SHIFT:  implicit kernel-3 im2col from [384][128] LDS activation buffer.
template <int KCHUNKS, bool SHIFT, int MODE>
__device__ __forceinline__ void conv_gemm(const _Float16* __restrict__ src,
                                          const _Float16* __restrict__ gwt,
                                          const float* __restrict__ gbias,
                                          _Float16* __restrict__ dst) {
  constexpr int KWID = KCHUNKS * 32;
  const int lane    = threadIdx.x & 31;
  const int wv      = threadIdx.x >> 5;
  const int halfSel = lane >> 4;   // lanes 16-31 hold the upper K half
  const int l15     = lane & 15;
  const int col     = wv * 16 + l15;    // output channel
  const float bcol  = gbias[col];

  // Hoist the whole B operand for this wave's column tile into VGPRs.
  v16h bfr[KCHUNKS];
#pragma unroll
  for (int kc = 0; kc < KCHUNKS; ++kc)
    bfr[kc] = *(const v16h*)(gwt + col * KWID + kc * 32 + halfSel * 16);

  for (int mt = 0; mt < MT; ++mt) {
    v8f acc = {};
    const int arow = mt * 16 + l15;       // A-matrix row (sample, pos)
    const int s    = arow / LBOARD;
    const int p    = arow - s * LBOARD;
#pragma unroll
    for (int kc = 0; kc < KCHUNKS; ++kc) {
      const int kbase = kc * 32 + halfSel * 16;
      v16h afrag = {};
      if (SHIFT) {
        const int dk   = kbase >> 7;      // tap 0..2
        const int c0   = kbase & (F - 1);
        const int psrc = p + dk - 1;      // SAME padding, kernel 3
        if (psrc >= 0 && psrc < LBOARD)
          afrag = *(const v16h*)(src + (s * LBOARD + psrc) * F + c0);
      } else {
        afrag = *(const v16h*)(src + arow * KWID + kbase);
      }
      acc = __builtin_amdgcn_wmma_f32_16x16x32_f16(
          false, afrag, false, bfr[kc], (short)0, acc, false, false);
    }
    // C/D layout: VGPR v -> M = v + 8*halfSel, N = lane&15
#pragma unroll
    for (int v = 0; v < 8; ++v) {
      const int r = mt * 16 + v + halfSel * 8;
      float val = acc[v] + bcol;
      if (MODE == 0) val = fmaxf(val, 0.0f);
      if (MODE == 2) {
        const float xo = (float)dst[r * F + col];
        dst[r * F + col] = (_Float16)(xo + val);
      } else {
        dst[r * F + col] = (_Float16)val;
      }
    }
  }
}

// Per-row LayerNorm(channels=128) + ReLU.
__device__ __forceinline__ void layernorm_relu(const _Float16* __restrict__ src,
                                               _Float16* __restrict__ dst,
                                               const float* __restrict__ sc,
                                               const float* __restrict__ bi) {
  for (int r = threadIdx.x; r < MROWS; r += THREADS) {
    float sum = 0.f, sq = 0.f;
    for (int c = 0; c < F; c += 8) {
      const v8h hv = *(const v8h*)(src + r * F + c);
#pragma unroll
      for (int j = 0; j < 8; ++j) {
        const float f = (float)hv[j];
        sum += f; sq += f * f;
      }
    }
    const float mu  = sum * (1.0f / F);
    const float var = sq * (1.0f / F) - mu * mu;
    const float inv = rsqrtf(var + 1e-6f);
    for (int c = 0; c < F; c += 8) {
      const v8h hv = *(const v8h*)(src + r * F + c);
      v8h ov;
#pragma unroll
      for (int j = 0; j < 8; ++j) {
        const float f = ((float)hv[j] - mu) * inv * sc[c + j] + bi[c + j];
        ov[j] = (_Float16)fmaxf(f, 0.0f);
      }
      *(v8h*)(dst + r * F + c) = ov;
    }
  }
}

// ---------------------------------------------------------------------------
extern "C" __global__ void __launch_bounds__(THREADS, 1)
value_net_fused(const float* __restrict__ board,  // (B,24,15)
                const float* __restrict__ aux,    // (B,6)
                const float* __restrict__ b0,     // (128)
                const float* __restrict__ ln1s,   // (9,128)
                const float* __restrict__ ln1b,
                const float* __restrict__ cb1,    // (9,128)
                const float* __restrict__ ln2s,
                const float* __restrict__ ln2b,
                const float* __restrict__ cb2,
                const float* __restrict__ dw,     // (134,64)
                const float* __restrict__ db,     // (64)
                const float* __restrict__ ow,     // (64,1)
                const float* __restrict__ ob,     // (1)
                const _Float16* __restrict__ wsw, // pre-transposed weights
                float* __restrict__ out) {
  extern __shared__ char smem[];
  _Float16* xb = (_Float16*)(smem + OFF_X);
  _Float16* ab = (_Float16*)(smem + OFF_A);
  _Float16* bb = (_Float16*)(smem + OFF_B);
  float* lns  = (float*)(smem + OFF_LNS);
  float* lnb  = (float*)(smem + OFF_LNB);
  float* pool = (float*)(smem + OFF_POOL);

  const int bbase = blockIdx.x * TILE_B;

  // ---- conv0: explicit im2col (zero-padded K=105->128) in LDS ----
  for (int idx = threadIdx.x; idx < MROWS * KW_C0; idx += THREADS) {
    const int row = idx >> 7;
    const int k   = idx & 127;
    float val = 0.0f;
    if (k < K0_REAL) {
      const int dk = k / CIN;
      const int ci = k - dk * CIN;
      const int s  = row / LBOARD;
      const int p  = row - s * LBOARD;
      const int ps = p + dk - 3;  // SAME padding, kernel 7
      if (ps >= 0 && ps < LBOARD)
        val = board[((bbase + s) * LBOARD + ps) * CIN + ci];
    }
    ab[row * KW_C0 + k] = (_Float16)val;
  }
  __syncthreads();

  conv_gemm<KCH_C0, false, 0>(ab, wsw, b0, xb);
  __syncthreads();

  // ---- 9 pre-activation residual blocks ----
  for (int blk = 0; blk < NB; ++blk) {
    const _Float16* wt1 = wsw + WS_C0 + (blk * 2 + 0) * WS_RES1;
    const _Float16* wt2 = wsw + WS_C0 + (blk * 2 + 1) * WS_RES1;

    for (int i = threadIdx.x; i < F; i += THREADS) {
      lns[i] = ln1s[blk * F + i];
      lnb[i] = ln1b[blk * F + i];
    }
    __builtin_prefetch(wt1 + (int)threadIdx.x * 64, 0, 1);
    __syncthreads();
    layernorm_relu(xb, ab, lns, lnb);
    __syncthreads();

    conv_gemm<KCH_RES, true, 1>(ab, wt1, cb1 + blk * F, bb);
    __syncthreads();

    for (int i = threadIdx.x; i < F; i += THREADS) {
      lns[i] = ln2s[blk * F + i];
      lnb[i] = ln2b[blk * F + i];
    }
    __builtin_prefetch(wt2 + (int)threadIdx.x * 64, 0, 1);
    __syncthreads();
    layernorm_relu(bb, ab, lns, lnb);
    __syncthreads();

    conv_gemm<KCH_RES, true, 2>(ab, wt2, cb2 + blk * F, xb);  // x += h
    __syncthreads();
  }

  // ---- global mean pool over positions + concat aux ----
  for (int idx = threadIdx.x; idx < TILE_B * F; idx += THREADS) {
    const int s = idx >> 7;
    const int c = idx & 127;
    float acc = 0.f;
#pragma unroll
    for (int p = 0; p < LBOARD; ++p) acc += (float)xb[(s * LBOARD + p) * F + c];
    pool[s * POOLW + c] = acc * (1.0f / LBOARD);
  }
  for (int idx = threadIdx.x; idx < TILE_B * 6; idx += THREADS) {
    const int s = idx / 6, c = idx - s * 6;
    pool[s * POOLW + F + c] = aux[(bbase + s) * 6 + c];
  }
  __syncthreads();

  // ---- head: wave per sample (2 samples/wave), lanes cover 64 hidden ----
  {
    const int lane = threadIdx.x & 31;
    const int wv   = threadIdx.x >> 5;
    for (int s = wv; s < TILE_B; s += 8) {
      float h0 = db[lane], h1 = db[lane + 32];
      for (int i = 0; i < F + 6; ++i) {
        const float v = pool[s * POOLW + i];
        h0 = fmaf(v, dw[i * 64 + lane], h0);
        h1 = fmaf(v, dw[i * 64 + lane + 32], h1);
      }
      h0 = fmaxf(h0, 0.f);
      h1 = fmaxf(h1, 0.f);
      float part = h0 * ow[lane] + h1 * ow[lane + 32];
#pragma unroll
      for (int off = 16; off > 0; off >>= 1) part += __shfl_down(part, off, 32);
      if (lane == 0) out[bbase + s] = tanhf(part + ob[0]) * 3.0f;
    }
  }
}

// ---------------------------------------------------------------------------
extern "C" void kernel_launch(void* const* d_in, const int* in_sizes, int n_in,
                              void* d_out, int out_size, void* d_ws,
                              size_t ws_size, hipStream_t stream) {
  (void)n_in; (void)out_size; (void)ws_size;
  const float* board = (const float*)d_in[0];
  const float* aux   = (const float*)d_in[1];
  const float* w0    = (const float*)d_in[2];
  const float* b0    = (const float*)d_in[3];
  const float* ln1s  = (const float*)d_in[4];
  const float* ln1b  = (const float*)d_in[5];
  const float* cw1   = (const float*)d_in[6];
  const float* cb1   = (const float*)d_in[7];
  const float* ln2s  = (const float*)d_in[8];
  const float* ln2b  = (const float*)d_in[9];
  const float* cw2   = (const float*)d_in[10];
  const float* cb2   = (const float*)d_in[11];
  const float* dw    = (const float*)d_in[12];
  const float* db    = (const float*)d_in[13];
  const float* ow    = (const float*)d_in[14];
  const float* ob    = (const float*)d_in[15];
  _Float16* wsw = (_Float16*)d_ws;

  const int B = in_sizes[0] / (LBOARD * CIN);
  const int grid = B / TILE_B;

  prep_weights<<<(WS_TOTAL + 255) / 256, 256, 0, stream>>>(w0, cw1, cw2, wsw);

  hipFuncSetAttribute((const void*)value_net_fused,
                      hipFuncAttributeMaxDynamicSharedMemorySize, SMEM_BYTES);
  value_net_fused<<<grid, THREADS, SMEM_BYTES, stream>>>(
      board, aux, b0, ln1s, ln1b, cb1, ln2s, ln2b, cb2,
      dw, db, ow, ob, wsw, (float*)d_out);
}
